// VLContrastHead_47931835023976
// MI455X (gfx1250) — matrix operand order repaired and checked by reading, not compiled, not run on hardware
//
#include <hip/hip_runtime.h>

// Problem constants (match reference)
#define B_SZ  8192
#define DV_SZ 1152
#define DT_SZ 768
#define D_SZ  1024
#define LN_EPS 1e-5f

typedef __attribute__((ext_vector_type(16))) __bf16 v16bf;
typedef __attribute__((ext_vector_type(8)))  float  v8f;
typedef __attribute__((ext_vector_type(4)))  unsigned int u32x4;
typedef __attribute__((ext_vector_type(8)))  unsigned int u32x8;

union FragBF {
    uint4 u[2];   // 32 bytes = 8 VGPRs
    v16bf v;
};

__device__ __forceinline__ unsigned short f32_to_bf16(float f) {
    union { float f; unsigned int u; } c;
    c.f = f;
    unsigned int u = c.u;
    unsigned int rnd = 0x7FFFu + ((u >> 16) & 1u);   // round-to-nearest-even
    u += rnd;
    return (unsigned short)(u >> 16);
}

// ---------------------------------------------------------------------------
// TDM: issue a 2D tile DMA (tile_dim0=32 bf16, tile_dim1=128 rows) from a
// row-major bf16 tensor (row stride = row_stride elements) into LDS.
// LDS layout: pad_interval=3 (64B) + pad_amount=3 (16B) -> row stride 80B,
// i.e. 40 bf16 = LDSW, matching the WMMA fragment addressing below.
// D# group0/group1 bitfields per CDNA5 ISA ch.8 (CDNA5 ISA).
// ---------------------------------------------------------------------------
__device__ __forceinline__ void tdm_load_tile_128x32_bf16(
    unsigned int lds_off_bytes, const void* gptr,
    unsigned int row_stride, unsigned int rows_total)
{
    unsigned long long ga = (unsigned long long)(uintptr_t)gptr;

    u32x4 g0;
    g0[0] = 1u;                                   // [1:0] count=1 (valid user D#)
    g0[1] = lds_off_bytes;                        // [63:32] lds_addr
    g0[2] = (unsigned int)ga;                     // [95:64] global_addr lo
    g0[3] = ((unsigned int)(ga >> 32) & 0x01FFFFFFu)  // [120:96] global_addr hi
          | (2u << 30);                           // [127:126] type=2 ("image")

    u32x8 g1;
    g1[0] = (1u << 16)        // data_size=1 -> 2 bytes (bf16)
          | (1u << 20)        // pad_enable
          | (3u << 22)        // pad_interval code 3 -> every 16 DWORDs (64B)
          | (3u << 25);       // pad_amount  code 3 -> 4 DWORDs (16B)
    g1[1] = (row_stride & 0xFFFFu) << 16;                         // tensor_dim0 lo16 @ [63:48]
    g1[2] = ((row_stride >> 16) & 0xFFFFu)                        // tensor_dim0 hi16
          | ((rows_total & 0xFFFFu) << 16);                       // tensor_dim1 lo16
    g1[3] = ((rows_total >> 16) & 0xFFFFu)                        // tensor_dim1 hi16
          | (32u << 16);                                          // tile_dim0 = 32
    g1[4] = 128u;                                                 // tile_dim1 = 128, tile_dim2 = 0
    g1[5] = row_stride;                                           // tensor_dim0_stride lo32
    g1[6] = 0u;                                                   // stride hi16 = 0, dim1_stride lo16 = 0
    g1[7] = 0u;                                                   // dim1_stride hi = 0

    asm volatile("tensor_load_to_lds %0, %1" :: "s"(g0), "s"(g1) : "memory");
}

// ---------------------------------------------------------------------------
// Kernel 1: per-row LayerNorm over Din, cast result to bf16 (modality LN)
// ---------------------------------------------------------------------------
__global__ __launch_bounds__(256) void ln_cast_kernel(
    const float* __restrict__ x, const float* __restrict__ g,
    const float* __restrict__ b, unsigned short* __restrict__ out, int Din)
{
    const int row = blockIdx.x;
    const float* xr = x + (size_t)row * Din;
    unsigned short* orow = out + (size_t)row * Din;

    float s = 0.f, s2 = 0.f;
    for (int i = threadIdx.x; i < Din; i += 256) {
        float v = xr[i];
        s += v; s2 += v * v;
    }
    __shared__ float rs[256];
    __shared__ float rq[256];
    rs[threadIdx.x] = s; rq[threadIdx.x] = s2;
    __syncthreads();
    for (int o = 128; o > 0; o >>= 1) {
        if (threadIdx.x < o) {
            rs[threadIdx.x] += rs[threadIdx.x + o];
            rq[threadIdx.x] += rq[threadIdx.x + o];
        }
        __syncthreads();
    }
    const float inv_n = 1.0f / (float)Din;
    const float mu  = rs[0] * inv_n;
    const float var = rq[0] * inv_n - mu * mu;
    const float inv = rsqrtf(var + LN_EPS);
    for (int i = threadIdx.x; i < Din; i += 256) {
        float v = (xr[i] - mu) * inv * g[i] + b[i];
        orow[i] = f32_to_bf16(v);
    }
}

// ---------------------------------------------------------------------------
// Kernel 2: elementwise f32 -> bf16 cast (for weights)
// ---------------------------------------------------------------------------
__global__ __launch_bounds__(256) void cast_bf16_kernel(
    const float* __restrict__ in, unsigned short* __restrict__ out, int n)
{
    int i = blockIdx.x * 256 + threadIdx.x;
    if (i < n) out[i] = f32_to_bf16(in[i]);
}

// ---------------------------------------------------------------------------
// Kernel 3: tiled bf16 WMMA GEMM (NT):  C[m,n] = sum_k A[m,k]*Bm[n,k] + epi
//   Block: 128 threads (4 waves), C tile 128x128, each wave 64x64.
//   K step 32. Operand tiles staged to LDS by the Tensor Data Mover
//   (tensor_load_to_lds, TENSORcnt), double-buffered: DMA of tile kk+32
//   overlaps the 16 v_wmma_f32_16x16x32_bf16 of tile kk.
//   epi == 0: C += biasv[n]                   (projection)
//   epi == 1: C = C*exp(*scale_p) + *bias_p   (logits)
// ---------------------------------------------------------------------------
#define LDSW 40                  // bf16 per LDS row (32 data + 8 pad) = 80B
#define TILE_SH (128 * LDSW)     // one tile in shorts
#define TILE_BY (TILE_SH * 2)    // one tile in bytes (10240)

__global__ __launch_bounds__(128) void gemm_bf16_wmma_kernel(
    const unsigned short* __restrict__ A,   // [M,K] bf16 row-major
    const unsigned short* __restrict__ Bm,  // [N,K] bf16 row-major
    float* __restrict__ C,                  // [M,N] f32 row-major
    const float* __restrict__ biasv,        // [N] (epi 0)
    const float* __restrict__ scale_p,      // scalar (epi 1)
    const float* __restrict__ bias_p,       // scalar (epi 1)
    int M, int N, int K, int epi)
{
    // [buf0.A][buf0.B][buf1.A][buf1.B]
    __shared__ __align__(16) unsigned short lds[4 * TILE_SH];

    const int tid  = threadIdx.x;
    const int lane = tid & 31;
    const int wave = tid >> 5;        // 0..3
    const int wm   = wave >> 1;       // wave row (0..1)
    const int wn   = wave & 1;        // wave col (0..1)
    const int l16  = lane & 15;
    const int lh   = lane >> 4;       // 0 or 1

    const int m0 = blockIdx.y * 128;
    const int n0 = blockIdx.x * 128;

    const unsigned int lds_base = (unsigned int)(uintptr_t)&lds[0]; // LDS byte offset

    v8f acc[4][4];
    #pragma unroll
    for (int i = 0; i < 4; ++i)
        #pragma unroll
        for (int j = 0; j < 4; ++j)
            acc[i][j] = (v8f){0.f,0.f,0.f,0.f,0.f,0.f,0.f,0.f};

    const unsigned short* gA0 = A  + (size_t)m0 * K;   // tile origin rows
    const unsigned short* gB0 = Bm + (size_t)n0 * K;

    // ---- prologue: DMA first K tile into buffer 0 (wave 0 drives the TDM) ----
    if (tid < 32) {
        tdm_load_tile_128x32_bf16(lds_base,           gA0, (unsigned)K, (unsigned)(M - m0));
        tdm_load_tile_128x32_bf16(lds_base + TILE_BY, gB0, (unsigned)K, (unsigned)(N - n0));
        __builtin_amdgcn_s_wait_tensorcnt(0);
    }
    __syncthreads();

    for (int kk = 0; kk < K; kk += 32) {
        const int cur = (kk >> 5) & 1;

        // ---- issue DMA for the next K tile into the other buffer ----
        if (kk + 32 < K && tid < 32) {
            const unsigned int nb = lds_base + (unsigned)(cur ^ 1) * (2u * TILE_BY);
            tdm_load_tile_128x32_bf16(nb,           gA0 + (kk + 32), (unsigned)K, (unsigned)(M - m0));
            tdm_load_tile_128x32_bf16(nb + TILE_BY, gB0 + (kk + 32), (unsigned)K, (unsigned)(N - n0));
        }

        const unsigned short* curA = lds + cur * (2 * TILE_SH);
        const unsigned short* curB = curA + TILE_SH;

        // ---- load fragments per ISA wave32 layouts ----
        FragBF a[4], b[4];
        #pragma unroll
        for (int i = 0; i < 4; ++i) {
            const unsigned short* base = curA + (wm * 64 + i * 16 + l16) * LDSW;
            // lanes 0-15: K 0..7 & 16..23 ; lanes 16-31: K 8..15 & 24..31
            a[i].u[0] = *reinterpret_cast<const uint4*>(base + lh * 8);
            a[i].u[1] = *reinterpret_cast<const uint4*>(base + 16 + lh * 8);
        }
        #pragma unroll
        for (int j = 0; j < 4; ++j) {
            const unsigned short* base = curB + (wn * 64 + j * 16 + l16) * LDSW;
            // lanes 0-15: K 0..7 & 16..23 ; lanes 16-31: K 8..15 & 24..31
            b[j].u[0] = *reinterpret_cast<const uint4*>(base + lh * 8);
            b[j].u[1] = *reinterpret_cast<const uint4*>(base + 16 + lh * 8);
        }

        #pragma unroll
        for (int i = 0; i < 4; ++i)
            #pragma unroll
            for (int j = 0; j < 4; ++j)
                acc[i][j] = __builtin_amdgcn_wmma_f32_16x16x32_bf16(
                    false, a[i].v, false, b[j].v, (short)0, acc[i][j],
                    false, false);

        // ---- DMA of next tile must land before anyone reads it ----
        if (tid < 32) __builtin_amdgcn_s_wait_tensorcnt(0);
        __syncthreads();
    }

    // ---- epilogue + store (C layout: VGPR r -> M = r + 8*lh, N = l16) ----
    float scale = 1.f, addc = 0.f;
    if (epi == 1) { scale = expf(scale_p[0]); addc = bias_p[0]; }

    #pragma unroll
    for (int i = 0; i < 4; ++i) {
        #pragma unroll
        for (int j = 0; j < 4; ++j) {
            const int n = n0 + wn * 64 + j * 16 + l16;
            const float bn = (epi == 0) ? biasv[n] : 0.f;
            const int mbase = m0 + wm * 64 + i * 16 + 8 * lh;
            #pragma unroll
            for (int r = 0; r < 8; ++r) {
                float v = acc[i][j][r];
                v = (epi == 0) ? (v + bn) : (v * scale + addc);
                C[(size_t)(mbase + r) * N + n] = v;
            }
        }
    }
}

// ---------------------------------------------------------------------------
// Kernel 4: shared LayerNorm (D=1024) in-place on f32 rows, plus
//           l2-normalized bf16 copy for the logits GEMM.
// ---------------------------------------------------------------------------
__global__ __launch_bounds__(256) void shared_ln_l2_kernel(
    float* __restrict__ y, const float* __restrict__ g,
    const float* __restrict__ b, unsigned short* __restrict__ nrm)
{
    const int row = blockIdx.x;
    float* yr = y + (size_t)row * D_SZ;
    unsigned short* nr = nrm + (size_t)row * D_SZ;

    float v[4];
    float s = 0.f, s2 = 0.f;
    #pragma unroll
    for (int r = 0; r < 4; ++r) {
        v[r] = yr[threadIdx.x + r * 256];
        s += v[r]; s2 += v[r] * v[r];
    }
    __shared__ float rs[256];
    __shared__ float rq[256];
    rs[threadIdx.x] = s; rq[threadIdx.x] = s2;
    __syncthreads();
    for (int o = 128; o > 0; o >>= 1) {
        if (threadIdx.x < o) {
            rs[threadIdx.x] += rs[threadIdx.x + o];
            rq[threadIdx.x] += rq[threadIdx.x + o];
        }
        __syncthreads();
    }
    const float mu  = rs[0] * (1.0f / D_SZ);
    const float var = rq[0] * (1.0f / D_SZ) - mu * mu;
    const float inv = rsqrtf(var + LN_EPS);

    float t[4];
    float ss = 0.f;
    #pragma unroll
    for (int r = 0; r < 4; ++r) {
        const int idx = threadIdx.x + r * 256;
        t[r] = (v[r] - mu) * inv * g[idx] + b[idx];
        ss += t[r] * t[r];
    }
    __syncthreads();
    rs[threadIdx.x] = ss;
    __syncthreads();
    for (int o = 128; o > 0; o >>= 1) {
        if (threadIdx.x < o) rs[threadIdx.x] += rs[threadIdx.x + o];
        __syncthreads();
    }
    const float l2 = sqrtf(rs[0]);
    const float rinv = 1.0f / fmaxf(l2, 1e-12f);
    #pragma unroll
    for (int r = 0; r < 4; ++r) {
        const int idx = threadIdx.x + r * 256;
        yr[idx] = t[r];
        nr[idx] = f32_to_bf16(t[r] * rinv);
    }
}

// ---------------------------------------------------------------------------
// Host launcher
// ---------------------------------------------------------------------------
extern "C" void kernel_launch(void* const* d_in, const int* in_sizes, int n_in,
                              void* d_out, int out_size, void* d_ws, size_t ws_size,
                              hipStream_t stream) {
    (void)in_sizes; (void)n_in; (void)out_size; (void)ws_size;

    const float* img_f  = (const float*)d_in[0];   // [B, DV]
    const float* txt_f  = (const float*)d_in[1];   // [B, DT]
    const float* Wv     = (const float*)d_in[2];   // [D, DV]
    const float* bv     = (const float*)d_in[3];   // [D]
    const float* Wt     = (const float*)d_in[4];   // [D, DT]
    const float* bt     = (const float*)d_in[5];   // [D]
    const float* ln_v_g = (const float*)d_in[6];
    const float* ln_v_b = (const float*)d_in[7];
    const float* ln_t_g = (const float*)d_in[8];
    const float* ln_t_b = (const float*)d_in[9];
    const float* ln_s_g = (const float*)d_in[10];
    const float* ln_s_b = (const float*)d_in[11];
    const float* logit_scale = (const float*)d_in[12];
    const float* logit_bias  = (const float*)d_in[13];
    // d_in[14] = compute_logits (always 1 in this harness)

    float* out        = (float*)d_out;
    float* out_img    = out;                                  // [B, D]
    float* out_txt    = out + (size_t)B_SZ * D_SZ;            // [B, D]
    float* out_logits = out + 2 * (size_t)B_SZ * D_SZ;        // [B, B]

    char* ws = (char*)d_ws;
    size_t off = 0;
    unsigned short* wsAv = (unsigned short*)(ws + off); off += (size_t)B_SZ * DV_SZ * 2;
    unsigned short* wsAt = (unsigned short*)(ws + off); off += (size_t)B_SZ * DT_SZ * 2;
    unsigned short* wsWv = (unsigned short*)(ws + off); off += (size_t)D_SZ * DV_SZ * 2;
    unsigned short* wsWt = (unsigned short*)(ws + off); off += (size_t)D_SZ * DT_SZ * 2;
    unsigned short* wsNi = (unsigned short*)(ws + off); off += (size_t)B_SZ * D_SZ * 2;
    unsigned short* wsNt = (unsigned short*)(ws + off); off += (size_t)B_SZ * D_SZ * 2;

    // 1) modality LayerNorm -> bf16
    ln_cast_kernel<<<B_SZ, 256, 0, stream>>>(img_f, ln_v_g, ln_v_b, wsAv, DV_SZ);
    ln_cast_kernel<<<B_SZ, 256, 0, stream>>>(txt_f, ln_t_g, ln_t_b, wsAt, DT_SZ);

    // 2) weight casts
    {
        int nv = D_SZ * DV_SZ, nt = D_SZ * DT_SZ;
        cast_bf16_kernel<<<(nv + 255) / 256, 256, 0, stream>>>(Wv, wsWv, nv);
        cast_bf16_kernel<<<(nt + 255) / 256, 256, 0, stream>>>(Wt, wsWt, nt);
    }

    // 3) projection GEMMs (WMMA + TDM): out = LN(x) @ W^T + b   -> into d_out
    {
        dim3 grid(D_SZ / 128, B_SZ / 128);   // (8, 64)
        gemm_bf16_wmma_kernel<<<grid, 128, 0, stream>>>(
            wsAv, wsWv, out_img, bv, nullptr, nullptr, B_SZ, D_SZ, DV_SZ, 0);
        gemm_bf16_wmma_kernel<<<grid, 128, 0, stream>>>(
            wsAt, wsWt, out_txt, bt, nullptr, nullptr, B_SZ, D_SZ, DT_SZ, 0);
    }

    // 4) shared LayerNorm in-place + l2-normalized bf16 copies
    shared_ln_l2_kernel<<<B_SZ, 256, 0, stream>>>(out_img, ln_s_g, ln_s_b, wsNi);
    shared_ln_l2_kernel<<<B_SZ, 256, 0, stream>>>(out_txt, ln_s_g, ln_s_b, wsNt);

    // 5) logits = l2(txt) @ l2(img)^T * exp(scale) + bias  (WMMA + TDM)
    {
        dim3 grid(B_SZ / 128, B_SZ / 128);   // (64, 64)
        gemm_bf16_wmma_kernel<<<grid, 128, 0, stream>>>(
            wsNt, wsNi, out_logits, nullptr, logit_scale, logit_bias,
            B_SZ, B_SZ, D_SZ, 1);
    }
}
